// CausalSelfAttention_16149077033607
// MI455X (gfx1250) — compile-verified, matrix operand order checked by reading, not run on hardware
//
#include <hip/hip_runtime.h>

typedef __bf16 bf16;
typedef __attribute__((ext_vector_type(16))) __bf16 v16bf;
typedef __attribute__((ext_vector_type(8)))  float  v8f;
typedef __attribute__((ext_vector_type(4)))  unsigned int u32x4;

#define BB 4
#define TT 2048
#define CC 1024
#define HH 16
#define DD 64
#define LDQKV (3 * CC)   // 3072

union FragU { v16bf bf; u32x4 u[2]; };

// Load a 16x32 bf16 WMMA fragment: this lane supplies row (row) of a row-major
// matrix with leading dim ld, K-offset k0.  Per ISA 16-bit A/B layout:
// lanes 0-15 hold K = {0..7, 16..23}, lanes 16-31 hold K = {8..15, 24..31}.
__device__ __forceinline__ v16bf load_frag(const bf16* __restrict__ base, int ld,
                                           int row, int k0, int lane) {
  const int kb = (lane & 16) ? 8 : 0;
  const bf16* p = base + (size_t)row * (size_t)ld + (size_t)(k0 + kb);
  FragU f;
  f.u[0] = *(const u32x4*)(p);
  f.u[1] = *(const u32x4*)(p + 16);
  return f.bf;
}

__device__ __forceinline__ v8f zero_v8f() {
  v8f z = {0.f, 0.f, 0.f, 0.f, 0.f, 0.f, 0.f, 0.f};
  return z;
}

// ---------------------------------------------------------------- conversion
__global__ void cvt_f32_to_bf16(const float* __restrict__ in,
                                bf16* __restrict__ out, long n, float scale) {
  long i = (long)blockIdx.x * blockDim.x + threadIdx.x;
  if (i < n) out[i] = (bf16)(in[i] * scale);
}

// --------------------------------------- V transpose: [B,T,(h d)] -> [B,H,D,T]
// LDS-tiled so both global reads and writes are coalesced.
// Block: 256 threads handles one (b, h, 32-row t tile) of the full 64-wide head.
__global__ __launch_bounds__(256) void transpose_v_kernel(
    const bf16* __restrict__ qkv, bf16* __restrict__ vt) {
  __shared__ bf16 tile[DD][34];  // [d][t], padded row

  const int tid = threadIdx.x;
  const int tt  = blockIdx.x & (TT / 32 - 1);  // 64 t-tiles
  const int h   = (blockIdx.x >> 6) & (HH - 1);
  const int b   = blockIdx.x >> 10;
  const int t0  = tt * 32;

  const bf16* src = qkv + ((size_t)b * TT + t0) * LDQKV + 2 * CC + h * DD;
  // Load 32 (t) x 64 (d): 1024 u32 reads of 2 bf16, coalesced along d.
#pragma unroll
  for (int it = 0; it < 4; ++it) {
    int idx = tid + it * 256;          // over 1024 u32
    int t = idx >> 5;                  // 32 u32 per t-row
    int c = idx & 31;                  // d pair
    union { unsigned int u; bf16 h2[2]; } sp;
    sp.u = *(const unsigned int*)(src + (size_t)t * LDQKV + 2 * c);
    tile[2 * c][t]     = sp.h2[0];
    tile[2 * c + 1][t] = sp.h2[1];
  }
  __syncthreads();
  // Store 64 (d) x 32 (t): 1024 u32 writes of 2 bf16, coalesced along t.
  bf16* dst = vt + ((size_t)(b * HH + h) * DD) * TT + t0;
#pragma unroll
  for (int it = 0; it < 4; ++it) {
    int idx = tid + it * 256;          // over 1024 u32
    int d = idx >> 4;                  // 16 u32 per d-row
    int tp = (idx & 15) * 2;
    union { unsigned int u; bf16 h2[2]; } sp;
    sp.h2[0] = tile[d][tp];
    sp.h2[1] = tile[d][tp + 1];
    *(unsigned int*)(dst + (size_t)d * TT + tp) = sp.u;
  }
}

// ------------------------------------------------------------- WMMA GEMM
// O[M,N] = A[M,K] * W[N,K]^T.  A,W row-major bf16.  Each wave: 32x64 tile.
template <bool OUT_F32>
__global__ __launch_bounds__(128) void wmma_gemm_kernel(
    const bf16* __restrict__ A, const bf16* __restrict__ W,
    void* __restrict__ out, int M, int N, int K) {
  const int lane = threadIdx.x & 31;
  const int wid  = threadIdx.x >> 5;
  const int wm = wid & 1, wn = wid >> 1;
  const int m0 = blockIdx.y * 64 + wm * 32;
  const int n0 = blockIdx.x * 128 + wn * 64;
  const int rl = lane & 15;

  v8f acc[2][4];
#pragma unroll
  for (int i = 0; i < 2; ++i)
#pragma unroll
    for (int j = 0; j < 4; ++j) acc[i][j] = zero_v8f();

  for (int k0 = 0; k0 < K; k0 += 32) {
    v16bf a[2], b[4];
#pragma unroll
    for (int mt = 0; mt < 2; ++mt)
      a[mt] = load_frag(A, K, m0 + mt * 16 + rl, k0, lane);
#pragma unroll
    for (int nt = 0; nt < 4; ++nt)
      b[nt] = load_frag(W, K, n0 + nt * 16 + rl, k0, lane);
#pragma unroll
    for (int mt = 0; mt < 2; ++mt)
#pragma unroll
      for (int nt = 0; nt < 4; ++nt)
        acc[mt][nt] = __builtin_amdgcn_wmma_f32_16x16x32_bf16(
            false, a[mt], false, b[nt], (short)0, acc[mt][nt], false, false);
  }

  const int mh = (lane >> 4) * 8;
#pragma unroll
  for (int mt = 0; mt < 2; ++mt)
#pragma unroll
    for (int nt = 0; nt < 4; ++nt)
#pragma unroll
      for (int v = 0; v < 8; ++v) {
        int row = m0 + mt * 16 + mh + v;
        int col = n0 + nt * 16 + rl;
        if (OUT_F32)
          ((float*)out)[(size_t)row * N + col] = acc[mt][nt][v];
        else
          ((bf16*)out)[(size_t)row * N + col] = (bf16)acc[mt][nt][v];
      }
}

// -------------------------------------------------------- flash attention
// One wave per 16-row q tile; 64 k-columns per softmax round; masking only in
// the final (diagonal) round.  Scale is pre-folded into the Q weights.
template <bool MASKED>
__device__ __forceinline__ void attn_round(
    int k0, int q0, int lane, int rl, int mh,
    const bf16* __restrict__ kbase, const bf16* __restrict__ vbase,
    const v16bf (&aQ)[2], v8f (&o)[4], float (&rm)[8], float (&rs)[8],
    bf16 (&pb)[16][72]) {
  // ---- S = Q K^T for 16 (q) x 64 (k): four 16x16 D tiles
  v8f s[4];
#pragma unroll
  for (int j = 0; j < 4; ++j) s[j] = zero_v8f();
#pragma unroll
  for (int kt = 0; kt < 2; ++kt)
#pragma unroll
    for (int j = 0; j < 4; ++j) {
      v16bf bk = load_frag(kbase, LDQKV, k0 + j * 16 + rl, kt * 32, lane);
      s[j] = __builtin_amdgcn_wmma_f32_16x16x32_bf16(false, aQ[kt], false, bk,
                                                     (short)0, s[j], false, false);
    }

  // ---- online softmax (slot v <-> row q0+mh+v; col groups k0 + 16j + rl)
#pragma unroll
  for (int v = 0; v < 8; ++v) {
    float x0 = s[0][v], x1 = s[1][v], x2 = s[2][v], x3 = s[3][v];
    if (MASKED) {
      const int gq = q0 + mh + v;
      if (k0 + rl      > gq) x0 = -3.0e38f;
      if (k0 + 16 + rl > gq) x1 = -3.0e38f;
      if (k0 + 32 + rl > gq) x2 = -3.0e38f;
      if (k0 + 48 + rl > gq) x3 = -3.0e38f;
    }
    float mx = fmaxf(fmaxf(x0, x1), fmaxf(x2, x3));
#pragma unroll
    for (int d = 1; d < 16; d <<= 1) mx = fmaxf(mx, __shfl_xor(mx, d, 32));
    const float nm = fmaxf(rm[v], mx);
    const float al = exp2f(rm[v] - nm);
    rm[v] = nm;
    const float e0 = exp2f(x0 - nm);
    const float e1 = exp2f(x1 - nm);
    const float e2 = exp2f(x2 - nm);
    const float e3 = exp2f(x3 - nm);
    float sm = (e0 + e1) + (e2 + e3);
#pragma unroll
    for (int d = 1; d < 16; d <<= 1) sm += __shfl_xor(sm, d, 32);
    rs[v] = rs[v] * al + sm;
#pragma unroll
    for (int nt = 0; nt < 4; ++nt) o[nt][v] *= al;
    pb[mh + v][rl]      = (bf16)e0;   // P[q][t] row-major in LDS
    pb[mh + v][rl + 16] = (bf16)e1;
    pb[mh + v][rl + 32] = (bf16)e2;
    pb[mh + v][rl + 48] = (bf16)e3;
  }

  // ---- read P back as two 16x32 A fragments (DS ops are in-order per wave)
  const int kb = (lane & 16) ? 8 : 0;
  const bf16* prow = &pb[rl][0];
  FragU p0, p1;
  p0.u[0] = *(const u32x4*)(prow + kb);
  p0.u[1] = *(const u32x4*)(prow + 16 + kb);
  p1.u[0] = *(const u32x4*)(prow + 32 + kb);
  p1.u[1] = *(const u32x4*)(prow + 48 + kb);

  // ---- O += P V : V^T rows are contiguous, so B frags are plain row loads
#pragma unroll
  for (int nt = 0; nt < 4; ++nt) {
    v16bf bv0 = load_frag(vbase, TT, nt * 16 + rl, k0, lane);
    v16bf bv1 = load_frag(vbase, TT, nt * 16 + rl, k0 + 32, lane);
    o[nt] = __builtin_amdgcn_wmma_f32_16x16x32_bf16(false, p0.bf, false, bv0,
                                                    (short)0, o[nt], false, false);
    o[nt] = __builtin_amdgcn_wmma_f32_16x16x32_bf16(false, p1.bf, false, bv1,
                                                    (short)0, o[nt], false, false);
  }
}

__global__ __launch_bounds__(128) void attn_kernel(
    const bf16* __restrict__ qkv, const bf16* __restrict__ vt,
    bf16* __restrict__ y) {
  __shared__ __align__(16) bf16 pbuf[4][16][72];  // per-wave P tile, padded

  const int lane = threadIdx.x & 31;
  const int wid  = threadIdx.x >> 5;
  const int tile = blockIdx.x * 4 + wid;   // B*H*(T/16) tiles
  const int qt = tile & 127;               // T/16 = 128
  const int bh = tile >> 7;
  const int h = bh & (HH - 1);
  const int b = bh >> 4;
  const int q0 = qt * 16;
  const int rl = lane & 15;
  const int mh = (lane >> 4) * 8;

  const bf16* qbase = qkv + (size_t)b * TT * LDQKV + h * DD;
  const bf16* kbase = qbase + CC;
  const bf16* vbase = vt + ((size_t)(b * HH + h) * DD) * TT;  // [d][t], ld = T

  v16bf aQ[2];
  aQ[0] = load_frag(qbase, LDQKV, q0 + rl, 0, lane);
  aQ[1] = load_frag(qbase, LDQKV, q0 + rl, 32, lane);

  v8f o[4];
#pragma unroll
  for (int nt = 0; nt < 4; ++nt) o[nt] = zero_v8f();
  float rm[8], rs[8];
#pragma unroll
  for (int v = 0; v < 8; ++v) { rm[v] = -3.0e38f; rs[v] = 0.f; }

  // k blocks: only the last 64-wide round can touch the causal diagonal
  // (k_last = 64*floor((q0+15)/64) <= q0), so earlier rounds need no mask.
  const int kLast = ((q0 + 15) >> 6) << 6;
  for (int k0 = 0; k0 < kLast; k0 += 64)
    attn_round<false>(k0, q0, lane, rl, mh, kbase, vbase, aQ, o, rm, rs,
                      pbuf[wid]);
  attn_round<true>(kLast, q0, lane, rl, mh, kbase, vbase, aQ, o, rm, rs,
                   pbuf[wid]);

  // ---- normalize and store y[b, t, h*64 + d] as bf16
#pragma unroll
  for (int v = 0; v < 8; ++v) {
    const float inv = 1.0f / rs[v];
    const int row = q0 + mh + v;
#pragma unroll
    for (int nt = 0; nt < 4; ++nt) {
      const int col = h * DD + nt * 16 + rl;
      y[((size_t)b * TT + row) * CC + col] = (bf16)(o[nt][v] * inv);
    }
  }
}

// ------------------------------------------------------------------ launch
extern "C" void kernel_launch(void* const* d_in, const int* in_sizes, int n_in,
                              void* d_out, int out_size, void* d_ws,
                              size_t ws_size, hipStream_t stream) {
  (void)in_sizes; (void)n_in; (void)out_size; (void)ws_size;
  const float* x      = (const float*)d_in[0];
  const float* w_attn = (const float*)d_in[1];
  const float* w_proj = (const float*)d_in[2];

  char* ws = (char*)d_ws;
  size_t off = 0;
  auto carve = [&](size_t bytes) {
    char* p = ws + off;
    off += (bytes + 255) & ~(size_t)255;
    return p;
  };

  bf16* xb   = (bf16*)carve((size_t)BB * TT * CC * 2);
  bf16* wab  = (bf16*)carve((size_t)3 * CC * CC * 2);
  bf16* wpb  = (bf16*)carve((size_t)CC * CC * 2);
  bf16* qkvb = (bf16*)carve((size_t)BB * TT * 3 * CC * 2);
  bf16* vtb  = (bf16*)carve((size_t)BB * TT * CC * 2);
  bf16* yb   = (bf16*)carve((size_t)BB * TT * CC * 2);

  const long nx  = (long)BB * TT * CC;   // 8,388,608
  const long nq  = (long)CC * CC;        // Q-weight block of w_attn
  const long nkv = (long)2 * CC * CC;    // K,V-weight block of w_attn
  const long nwp = (long)CC * CC;

  const float qscale = 0.125f * 1.44269504088896f;  // 1/sqrt(64) * log2(e)

  cvt_f32_to_bf16<<<(nx + 255) / 256, 256, 0, stream>>>(x, xb, nx, 1.0f);
  // rows [0,C) of w_attn produce Q: fold softmax scale (and log2 e) in here.
  cvt_f32_to_bf16<<<(nq + 255) / 256, 256, 0, stream>>>(w_attn, wab, nq, qscale);
  cvt_f32_to_bf16<<<(nkv + 255) / 256, 256, 0, stream>>>(w_attn + nq, wab + nq,
                                                         nkv, 1.0f);
  cvt_f32_to_bf16<<<(nwp + 255) / 256, 256, 0, stream>>>(w_proj, wpb, nwp, 1.0f);

  // qkv = x @ w_attn^T  -> bf16 [B*T, 3C]
  wmma_gemm_kernel<false><<<dim3(3 * CC / 128, BB * TT / 64), 128, 0, stream>>>(
      xb, wab, (void*)qkvb, BB * TT, 3 * CC, CC);

  // V -> [B,H,D,T] (coalesced LDS-tiled transpose)
  transpose_v_kernel<<<BB * HH * (TT / 32), 256, 0, stream>>>(qkvb, vtb);

  attn_kernel<<<(BB * HH * (TT / 16)) / 4, 128, 0, stream>>>(qkvb, vtb, yb);

  // out = y @ w_proj^T -> f32 [B*T, C]
  wmma_gemm_kernel<true><<<dim3(CC / 128, BB * TT / 64), 128, 0, stream>>>(
      yb, wpb, d_out, BB * TT, CC, CC);
}
